// SynthesizerRandomSelfAttention_70772471103578
// MI455X (gfx1250) — matrix operand order repaired and checked by reading, not compiled, not run on hardware
//
#include <hip/hip_runtime.h>
#include <hip/hip_bf16.h>

typedef __bf16 bf16_t;
typedef __attribute__((ext_vector_type(16))) __bf16 v16bf;
typedef __attribute__((ext_vector_type(8)))  __bf16 v8bf;
typedef __attribute__((ext_vector_type(8)))  float  v8f;

#define WMMA_BF16(a, b, c) \
  __builtin_amdgcn_wmma_f32_16x16x32_bf16(false, (a), false, (b), (short)0, (c), false, false)

// ---------------------------------------------------------------------------
// Fragment loaders (wave32; contiguous k order used alike for A and B, not the ISA table order)
// A 16x32 bf16: lane l holds row m0+(l&15); K = h*16 + j, j=0..15, h=l>>4
// ---------------------------------------------------------------------------
static __device__ __forceinline__ v16bf ld_a(const bf16_t* A, int lda, int m0, int k0) {
  const int l = threadIdx.x & 31;
  const int m = m0 + (l & 15);
  const int h = l >> 4;
  const bf16_t* p = A + (size_t)m * lda + (k0 + h * 16);
  v8bf lo = *(const v8bf*)p;
  v8bf hi = *(const v8bf*)(p + 8);
  v16bf r;
#pragma unroll
  for (int i = 0; i < 8; ++i) { r[i] = lo[i]; r[i + 8] = hi[i]; }
  return r;
}

// B 32x16 bf16 from BT row-major [N][K]: lane l = col n0+(l&15),
// K = k0 + (l>>4)*16 + j, j=0..15  (16 contiguous bf16 per lane)
static __device__ __forceinline__ v16bf ld_bt(const bf16_t* BT, int ldb, int n0, int k0) {
  const int l = threadIdx.x & 31;
  const int n = n0 + (l & 15);
  const bf16_t* p = BT + (size_t)n * ldb + k0 + ((l >> 4) << 4);
  v8bf lo = *(const v8bf*)p;
  v8bf hi = *(const v8bf*)(p + 8);
  v16bf r;
#pragma unroll
  for (int i = 0; i < 8; ++i) { r[i] = lo[i]; r[i + 8] = hi[i]; }
  return r;
}

// D 16x16 f32: lane l = col n0+(l&15), rows m0+(l>>4)*8 + r
static __device__ __forceinline__ void st_rm_f32(float* D, int ldd, int m0, int n0, v8f c) {
  const int l = threadIdx.x & 31;
  const int n = n0 + (l & 15);
  const int mb = m0 + ((l >> 4) << 3);
#pragma unroll
  for (int r = 0; r < 8; ++r) D[(size_t)(mb + r) * ldd + n] = c[r];
}

// transposed bf16 store: DT[n][m]; 8 consecutive elems per lane -> one b128 store
static __device__ __forceinline__ void st_tr_bf16(bf16_t* DT, int ldd, int m0, int n0, v8f c) {
  const int l = threadIdx.x & 31;
  const int n = n0 + (l & 15);
  const int mb = m0 + ((l >> 4) << 3);
  v8bf o;
#pragma unroll
  for (int r = 0; r < 8; ++r) o[r] = (bf16_t)c[r];
  *(v8bf*)(DT + (size_t)n * ldd + mb) = o;
}

// ---------------------------------------------------------------------------
// Pre-processing kernels
// ---------------------------------------------------------------------------
__global__ void k_cvt_bf16(const float* __restrict__ s, bf16_t* __restrict__ d, int n) {
  int i = blockIdx.x * blockDim.x + threadIdx.x;
  if (i < n) d[i] = (bf16_t)s[i];
}

// x_mid [b][c][y][x] fp32 -> [b][y*64+x][c] bf16 (per-batch 2^20 elems)
__global__ void k_nchw_to_nhwc(const float* __restrict__ s, bf16_t* __restrict__ d) {
  int i = blockIdx.x * blockDim.x + threadIdx.x;
  int b = i >> 20;
  int rem = i & ((1 << 20) - 1);
  int n = rem >> 8;
  int c = rem & 255;
  d[i] = (bf16_t)s[((size_t)b << 20) + ((size_t)c << 12) + n];
}

// g_w [co][ci][ky][kx] fp32 -> gwr [co][tap*256+ci] bf16
__global__ void k_gw_reorder(const float* __restrict__ s, bf16_t* __restrict__ d) {
  int i = blockIdx.x * blockDim.x + threadIdx.x;   // 589824 total
  int co = i / 2304;
  int k  = i - co * 2304;
  int tap = k >> 8, ci = k & 255;
  int ky = tap / 3, kx = tap - ky * 3;
  d[i] = (bf16_t)s[((size_t)(co * 256 + ci)) * 9 + ky * 3 + kx];
}

// random_map[c][d] = sum_r w1[c][r]*w2[d][r]
__global__ void k_rm(const float* __restrict__ w1, const float* __restrict__ w2,
                     float* __restrict__ rm) {
  int i = blockIdx.x * blockDim.x + threadIdx.x;   // 65536
  int c = i >> 8, d = i & 255;
  float s = 0.f;
#pragma unroll
  for (int r = 0; r < 16; ++r) s += w1[c * 16 + r] * w2[d * 16 + r];
  rm[i] = s;
}

// ---------------------------------------------------------------------------
// WMMA GEMM kernels: 256 thr = 8 waves (2x4); wave tile = 32x64 (2x4 WMMA accs)
// block tile = 64(M) x 256(N)
// ---------------------------------------------------------------------------

// valueT[b][n][co] = conv3x3(x_mid, g_w):  M=256(co) N=4096 K=2304
// The wave's 64 columns cover exactly one image row (n0 % 64 == 0), so yy and
// the row base are wave-uniform per tap; only 32-bit x-offsets are per-lane.
__global__ void __launch_bounds__(256, 1)
k_conv3x3(const bf16_t* __restrict__ gwr, const bf16_t* __restrict__ xm,
          bf16_t* __restrict__ valT) {
  const int wave = threadIdx.x >> 5;
  const int m0 = blockIdx.y * 64 + (wave >> 2) * 32;
  const int n0 = blockIdx.x * 256 + (wave & 3) * 64;   // multiple of 64
  const int l = threadIdx.x & 31;
  const int lx = l & 15;
  const int y = n0 >> 6;                               // uniform image row
  const int koff = (l >> 4) << 4;                      // ci sub-offset of B frag
  const bf16_t* xmb = xm + ((size_t)blockIdx.z << 20) + koff;

  v8f acc[2][4] = {};
#pragma unroll 1
  for (int ky = 0; ky < 3; ++ky) {
    const int yy = y + ky - 1;
    const bool oky = (unsigned)yy < 64u;               // wave-uniform
    const bf16_t* rowp = xmb + (yy << 14);             // row base (guarded by oky)
#pragma unroll 1
    for (int kx = 0; kx < 3; ++kx) {
      const int kbase = (ky * 3 + kx) << 8;
      int xoff[4];
      bool okx[4];
#pragma unroll
      for (int j = 0; j < 4; ++j) {
        int xx = j * 16 + lx + kx - 1;
        okx[j] = oky & ((unsigned)xx < 64u);
        xoff[j] = xx << 8;
      }
#pragma unroll 1
      for (int c0 = 0; c0 < 256; c0 += 32) {
        v16bf a[2];
        a[0] = ld_a(gwr, 2304, m0,      kbase + c0);
        a[1] = ld_a(gwr, 2304, m0 + 16, kbase + c0);
        v16bf b[4];
#pragma unroll
        for (int j = 0; j < 4; ++j) {
          v16bf r = {};
          if (okx[j]) {
            const bf16_t* p = rowp + xoff[j] + c0;
            v8bf lo = *(const v8bf*)p;
            v8bf hi = *(const v8bf*)(p + 8);
#pragma unroll
            for (int i = 0; i < 8; ++i) { r[i] = lo[i]; r[i + 8] = hi[i]; }
          }
          b[j] = r;
        }
#pragma unroll
        for (int i = 0; i < 2; ++i)
#pragma unroll
          for (int j = 0; j < 4; ++j)
            acc[i][j] = WMMA_BF16(a[i], b[j], acc[i][j]);
      }
    }
  }
  bf16_t* D = valT + ((size_t)blockIdx.z << 20);
#pragma unroll
  for (int i = 0; i < 2; ++i)
#pragma unroll
    for (int j = 0; j < 4; ++j)
      st_tr_bf16(D, 256, m0 + 16 * i, n0 + 16 * j, acc[i][j]);
}

// sim[b][c][d] = X Xt : M=N=256, K=4096 (A and BT are the same row-major X)
__global__ void __launch_bounds__(256, 1)
k_sim(const bf16_t* __restrict__ xb, float* __restrict__ sim) {
  const int wave = threadIdx.x >> 5;
  const int m0 = (wave >> 2) * 32 + blockIdx.y * 64;
  const int n0 = (wave & 3) * 64;
  const bf16_t* X = xb + ((size_t)blockIdx.z << 20);
  v8f acc[2][4] = {};
#pragma unroll 1
  for (int k0 = 0; k0 < 4096; k0 += 32) {
    v16bf a[2];
    a[0] = ld_a(X, 4096, m0,      k0);
    a[1] = ld_a(X, 4096, m0 + 16, k0);
    v16bf b[4];
#pragma unroll
    for (int j = 0; j < 4; ++j) b[j] = ld_bt(X, 4096, n0 + 16 * j, k0);
#pragma unroll
    for (int i = 0; i < 2; ++i)
#pragma unroll
      for (int j = 0; j < 4; ++j)
        acc[i][j] = WMMA_BF16(a[i], b[j], acc[i][j]);
  }
  float* D = sim + ((size_t)blockIdx.z << 16);
#pragma unroll
  for (int i = 0; i < 2; ++i)
#pragma unroll
    for (int j = 0; j < 4; ++j)
      st_rm_f32(D, 256, m0 + 16 * i, n0 + 16 * j, acc[i][j]);
}

// attn[b][c][d] = softmax(sim/16 + rm) -> bf16 row-major
__global__ void k_softmax(const float* __restrict__ sim, const float* __restrict__ rm,
                          bf16_t* __restrict__ attn) {
  const int row = blockIdx.x;            // b*256 + c
  const int c = row & 255;
  const int d = threadIdx.x;
  float v = sim[((size_t)row << 8) + d] * 0.0625f + rm[(c << 8) + d];
  __shared__ float red[256];
  red[d] = v; __syncthreads();
  for (int s = 128; s > 0; s >>= 1) { if (d < s) red[d] = fmaxf(red[d], red[d + s]); __syncthreads(); }
  const float mx = red[0]; __syncthreads();
  const float e = __expf(v - mx);
  red[d] = e; __syncthreads();
  for (int s = 128; s > 0; s >>= 1) { if (d < s) red[d] += red[d + s]; __syncthreads(); }
  attn[((size_t)row << 8) + d] = (bf16_t)(e / red[0]);
}

// contextT[b][n][c] = attn @ value : M=256(c) N=4096 K=256
__global__ void __launch_bounds__(256, 1)
k_ctx(const bf16_t* __restrict__ attn, const bf16_t* __restrict__ valT,
      bf16_t* __restrict__ ctxT) {
  const int wave = threadIdx.x >> 5;
  const int m0 = blockIdx.y * 64 + (wave >> 2) * 32;
  const int n0 = blockIdx.x * 256 + (wave & 3) * 64;
  const bf16_t* A = attn + ((size_t)blockIdx.z << 16);
  const bf16_t* B = valT + ((size_t)blockIdx.z << 20);
  v8f acc[2][4] = {};
#pragma unroll 1
  for (int k0 = 0; k0 < 256; k0 += 32) {
    v16bf a[2];
    a[0] = ld_a(A, 256, m0,      k0);
    a[1] = ld_a(A, 256, m0 + 16, k0);
    v16bf b[4];
#pragma unroll
    for (int j = 0; j < 4; ++j) b[j] = ld_bt(B, 256, n0 + 16 * j, k0);
#pragma unroll
    for (int i = 0; i < 2; ++i)
#pragma unroll
      for (int j = 0; j < 4; ++j)
        acc[i][j] = WMMA_BF16(a[i], b[j], acc[i][j]);
  }
  bf16_t* D = ctxT + ((size_t)blockIdx.z << 20);
#pragma unroll
  for (int i = 0; i < 2; ++i)
#pragma unroll
    for (int j = 0; j < 4; ++j)
      st_tr_bf16(D, 256, m0 + 16 * i, n0 + 16 * j, acc[i][j]);
}

// out[b][co][n] = w_w @ context : M=256 N=4096 K=256, fp32 output
__global__ void __launch_bounds__(256, 1)
k_out(const bf16_t* __restrict__ wb, const bf16_t* __restrict__ ctxT,
      float* __restrict__ out) {
  const int wave = threadIdx.x >> 5;
  const int m0 = blockIdx.y * 64 + (wave >> 2) * 32;
  const int n0 = blockIdx.x * 256 + (wave & 3) * 64;
  const bf16_t* B = ctxT + ((size_t)blockIdx.z << 20);
  v8f acc[2][4] = {};
#pragma unroll 1
  for (int k0 = 0; k0 < 256; k0 += 32) {
    v16bf a[2];
    a[0] = ld_a(wb, 256, m0,      k0);
    a[1] = ld_a(wb, 256, m0 + 16, k0);
    v16bf b[4];
#pragma unroll
    for (int j = 0; j < 4; ++j) b[j] = ld_bt(B, 256, n0 + 16 * j, k0);
#pragma unroll
    for (int i = 0; i < 2; ++i)
#pragma unroll
      for (int j = 0; j < 4; ++j)
        acc[i][j] = WMMA_BF16(a[i], b[j], acc[i][j]);
  }
  float* D = out + ((size_t)blockIdx.z << 20);
#pragma unroll
  for (int i = 0; i < 2; ++i)
#pragma unroll
    for (int j = 0; j < 4; ++j)
      st_rm_f32(D, 4096, m0 + 16 * i, n0 + 16 * j, acc[i][j]);
}

// ---------------------------------------------------------------------------
extern "C" void kernel_launch(void* const* d_in, const int* in_sizes, int n_in,
                              void* d_out, int out_size, void* d_ws, size_t ws_size,
                              hipStream_t stream) {
  const float* x     = (const float*)d_in[0];
  const float* x_mid = (const float*)d_in[1];
  const float* g_w   = (const float*)d_in[2];
  const float* w_w   = (const float*)d_in[3];
  const float* w1    = (const float*)d_in[4];
  const float* w2    = (const float*)d_in[5];
  float* out = (float*)d_out;

  char* ws = (char*)d_ws;
  bf16_t* xb   = (bf16_t*)(ws);                 //  67,108,864 B : x bf16 [b][c][n]
  bf16_t* xm   = (bf16_t*)(ws +  67108864);     //  67,108,864 B : x_mid NHWC bf16
  bf16_t* valT = (bf16_t*)(ws + 134217728);     //  67,108,864 B : value^T [b][n][c]
  bf16_t* gwr  = (bf16_t*)(ws + 201326592);     //   1,179,648 B : reordered conv3 wts
  bf16_t* wb   = (bf16_t*)(ws + 202506240);     //     131,072 B : 1x1 wts bf16
  float*  rm   = (float*) (ws + 202637312);     //     262,144 B : low-rank bias
  float*  sim  = (float*) (ws + 202899456);     //   8,388,608 B : sim fp32
  bf16_t* attn = (bf16_t*)(ws + 211288064);     //   4,194,304 B : attn bf16
  bf16_t* ctxT = (bf16_t*)(ws);                 //  reuse xb region (dead after k_sim)

  const int NE = 32 * 256 * 4096;               // 33,554,432

  k_cvt_bf16    <<<NE / 256, 256, 0, stream>>>(x, xb, NE);
  k_nchw_to_nhwc<<<NE / 256, 256, 0, stream>>>(x_mid, xm);
  k_gw_reorder  <<<2304, 256, 0, stream>>>(g_w, gwr);
  k_cvt_bf16    <<<256, 256, 0, stream>>>(w_w, wb, 65536);
  k_rm          <<<256, 256, 0, stream>>>(w1, w2, rm);

  k_conv3x3 <<<dim3(16, 4, 32), 256, 0, stream>>>(gwr, xm, valT);
  k_sim     <<<dim3( 1, 4, 32), 256, 0, stream>>>(xb, sim);
  k_softmax <<<8192, 256, 0, stream>>>(sim, rm, attn);
  k_ctx     <<<dim3(16, 4, 32), 256, 0, stream>>>(attn, valT, ctxT);
  k_out     <<<dim3(16, 4, 32), 256, 0, stream>>>(wb, ctxT, out);
}
